// MambaModel_80539226734847
// MI455X (gfx1250) — matrix operand order repaired and checked by reading, not compiled, not run on hardware
//
#include <hip/hip_runtime.h>
#include <hip/hip_bf16.h>
#include <math.h>

// ---------------- types ----------------
typedef __attribute__((ext_vector_type(16))) __bf16 v16bf;
typedef __attribute__((ext_vector_type(8)))  __bf16 v8bf;
typedef __attribute__((ext_vector_type(8)))  float  v8f;
typedef int v4i_ __attribute__((vector_size(16)));   // matches async-LDS builtin param

#define SHUF16(lo, hi) __builtin_shufflevector(lo, hi, 0,1,2,3,4,5,6,7,8,9,10,11,12,13,14,15)

// ---------------- model dims ----------------
constexpr int kB    = 8;
constexpr int kL    = 1024;
constexpr int kBL   = kB * kL;        // 8192 rows
constexpr int kMels = 128;
constexpr int kDM   = 512;
constexpr int kDI   = 1024;           // d_inner
constexpr int kDS   = 32;             // d_state
constexpr int kDTR  = 32;             // dt_rank
constexpr int kDC   = 4;              // d_conv
constexpr int kNL   = 2;
constexpr int kNC   = 1251;

constexpr int ACT_NONE = 0, ACT_SOFTPLUS = 1, ACT_SILU = 2;

#if __has_builtin(__builtin_amdgcn_global_load_async_to_lds_b128)
#define HAVE_ASYNC_LDS 1
#endif

// ---------------- helpers ----------------
__device__ __forceinline__ __bf16 f2bf(float f) {
  union { float f; unsigned u; } in; in.f = f;
  unsigned r = in.u + 0x7FFFu + ((in.u >> 16) & 1u);   // RNE
  union { unsigned short s; __bf16 b; } out;
  out.s = (unsigned short)(r >> 16);
  return out.b;
}

__device__ __forceinline__ float silu_f(float x) { return x / (1.0f + expf(-x)); }

__device__ __forceinline__ float act_apply(float v, int act) {
  if (act == ACT_SOFTPLUS) return (v > 20.0f) ? v : log1pf(expf(v));
  if (act == ACT_SILU)     return silu_f(v);
  return v;
}

// 16B global -> LDS stage (async CDNA5 path when toolchain exposes it)
__device__ __forceinline__ void stage16(const __bf16* g, __bf16* l) {
#if defined(HAVE_ASYNC_LDS)
  // builtin signature: (v4i addrspace(1)*, v4i addrspace(3)*, imm offset, imm cpol)
  __attribute__((address_space(1))) v4i_* gp =
      (__attribute__((address_space(1))) v4i_*)(unsigned long long)g;
  __attribute__((address_space(3))) v4i_* lp =
      (__attribute__((address_space(3))) v4i_*)(unsigned int)(unsigned long long)l;
  __builtin_amdgcn_global_load_async_to_lds_b128(gp, lp, 0, 0);
#else
  *(v8bf*)l = *(const v8bf*)g;
#endif
}

__device__ __forceinline__ void wait_stage() {
#if defined(HAVE_ASYNC_LDS)
#if __has_builtin(__builtin_amdgcn_s_wait_asynccnt)
  __builtin_amdgcn_s_wait_asynccnt(0);
#else
  asm volatile("s_wait_asynccnt 0" ::: "memory");
#endif
#endif
}

// ---------------- f32 -> bf16 convert (weights + input x) ----------------
__global__ __launch_bounds__(256) void f32_to_bf16_kernel(
    const float* __restrict__ src, __bf16* __restrict__ dst, int n) {
  int i = blockIdx.x * 256 + threadIdx.x;
  if (i < n) dst[i] = f2bf(src[i]);
}

// ---------------- epilogue store (f32 + optional bf16 mirror) ----------------
__device__ __forceinline__ void store_tile(float* __restrict__ C, __bf16* Cb, int ldc,
                                           int row0, int col0, int lane,
                                           v8f acc, const float* __restrict__ bias,
                                           int act) {
  const int n   = lane & 15;
  const int mb  = (lane >> 4) * 8;
  const int col = col0 + n;
  const float bv = bias ? bias[col] : 0.0f;
#pragma unroll
  for (int r = 0; r < 8; ++r) {
    float v = acc[r] + bv;
    v = act_apply(v, act);
    size_t idx = (size_t)(row0 + mb + r) * ldc + col;
    C[idx] = v;
    if (Cb) Cb[idx] = f2bf(v);
  }
}

// ================= direct WMMA GEMM (odd shapes) =================
// one wave computes a 32x32 tile; A[M,K] bf16 (lda), W[N,K] bf16 row-major
__global__ __launch_bounds__(32) void wmma_gemm_direct(
    const __bf16* __restrict__ A, int lda,
    const __bf16* __restrict__ W,
    const float*  __restrict__ bias,
    float* __restrict__ C, __bf16* Cb, int N, int K, int act) {
  const int lane = threadIdx.x;
  const int row0 = blockIdx.x * 32;
  const int col0 = blockIdx.y * 32;
  const int ln   = lane & 15;
  const int hf   = lane >> 4;

  v8f acc00 = {0,0,0,0,0,0,0,0};
  v8f acc01 = acc00, acc10 = acc00, acc11 = acc00;

  const __bf16* a0 = A + (size_t)(row0 + ln) * lda;
  const __bf16* a1 = a0 + (size_t)16 * lda;
  const __bf16* w0 = W + (size_t)(col0 + ln) * K;
  const __bf16* w1 = w0 + (size_t)16 * K;

  for (int k0 = 0; k0 < K; k0 += 32) {
    v8bf alo0 = *(const v8bf*)(a0 + k0 + 8 * hf);
    v8bf ahi0 = *(const v8bf*)(a0 + k0 + 16 + 8 * hf);
    v8bf alo1 = *(const v8bf*)(a1 + k0 + 8 * hf);
    v8bf ahi1 = *(const v8bf*)(a1 + k0 + 16 + 8 * hf);
    v16bf fa0 = SHUF16(alo0, ahi0);
    v16bf fa1 = SHUF16(alo1, ahi1);
    v8bf blo0 = *(const v8bf*)(w0 + k0 + 8 * hf);
    v8bf bhi0 = *(const v8bf*)(w0 + k0 + 16 + 8 * hf);
    v8bf blo1 = *(const v8bf*)(w1 + k0 + 8 * hf);
    v8bf bhi1 = *(const v8bf*)(w1 + k0 + 16 + 8 * hf);
    v16bf fb0 = SHUF16(blo0, bhi0);
    v16bf fb1 = SHUF16(blo1, bhi1);

    acc00 = __builtin_amdgcn_wmma_f32_16x16x32_bf16(false, fa0, false, fb0, (short)0, acc00, false, false);
    acc01 = __builtin_amdgcn_wmma_f32_16x16x32_bf16(false, fa0, false, fb1, (short)0, acc01, false, false);
    acc10 = __builtin_amdgcn_wmma_f32_16x16x32_bf16(false, fa1, false, fb0, (short)0, acc10, false, false);
    acc11 = __builtin_amdgcn_wmma_f32_16x16x32_bf16(false, fa1, false, fb1, (short)0, acc11, false, false);
  }

  store_tile(C, Cb, N, row0 +  0, col0 +  0, lane, acc00, bias, act);
  store_tile(C, Cb, N, row0 +  0, col0 + 16, lane, acc01, bias, act);
  store_tile(C, Cb, N, row0 + 16, col0 +  0, lane, acc10, bias, act);
  store_tile(C, Cb, N, row0 + 16, col0 + 16, lane, acc11, bias, act);
}

// ================= LDS-staged WMMA GEMM, double-buffered async refill =================
// 256 threads, block tile 128x128; waves 4(M)x2(N): each wave 32x64 (2x4 WMMA tiles)
// requires: lda == K, M % 128 == 0, N % 128 == 0, K % 32 == 0
constexpr int kRS = 40;  // LDS row stride in halves (32 data + 8 pad) = 80 B

__global__ __launch_bounds__(256) void wmma_gemm_lds(
    const __bf16* __restrict__ A,   // [M,K]
    const __bf16* __restrict__ W,   // [N,K]
    const float*  __restrict__ bias,
    float* __restrict__ C, __bf16* Cb, int N, int K, int act) {
  __shared__ __bf16 lA[2][128 * kRS];
  __shared__ __bf16 lB[2][128 * kRS];

  const int t    = threadIdx.x;
  const int lane = t & 31;
  const int w    = t >> 5;
  const int wm   = w >> 1;            // 0..3
  const int wn   = w & 1;             // 0..1
  const int ln   = lane & 15;
  const int hf   = lane >> 4;

  const int rowBase = blockIdx.x * 128;
  const int colBase = blockIdx.y * 128;

  v8f acc[2][4];
#pragma unroll
  for (int i = 0; i < 2; ++i)
#pragma unroll
    for (int j = 0; j < 4; ++j) acc[i][j] = v8f{0,0,0,0,0,0,0,0};

  // this thread's two 16B staging chunks: chunk c -> row c>>2, kpart c&3
  const int c0r = (t + 0)   >> 2, c0k = (t + 0)   & 3;
  const int c1r = (t + 256) >> 2, c1k = (t + 256) & 3;

#define STAGE_ALL(k0, buf)                                                              \
  do {                                                                                  \
    stage16(A + (size_t)(rowBase + c0r) * K + (k0) + c0k * 8, &lA[buf][c0r * kRS + c0k * 8]); \
    stage16(A + (size_t)(rowBase + c1r) * K + (k0) + c1k * 8, &lA[buf][c1r * kRS + c1k * 8]); \
    stage16(W + (size_t)(colBase + c0r) * K + (k0) + c0k * 8, &lB[buf][c0r * kRS + c0k * 8]); \
    stage16(W + (size_t)(colBase + c1r) * K + (k0) + c1k * 8, &lB[buf][c1r * kRS + c1k * 8]); \
  } while (0)

  const int nsteps = K >> 5;
  STAGE_ALL(0, 0);

  for (int s = 0; s < nsteps; ++s) {
    const int buf = s & 1;
    wait_stage();          // stage s landed (this wave's asyncs)
    __syncthreads();       // all waves' stage-s writes visible; all done reading buf^1
    if (s + 1 < nsteps) STAGE_ALL((s + 1) << 5, buf ^ 1);   // overlaps with compute below

    v16bf fa[2];
#pragma unroll
    for (int i = 0; i < 2; ++i) {
      const __bf16* la = &lA[buf][(wm * 32 + i * 16 + ln) * kRS];
      v8bf lo = *(const v8bf*)(la + 8 * hf);
      v8bf hi = *(const v8bf*)(la + 16 + 8 * hf);
      fa[i] = SHUF16(lo, hi);
    }
    v16bf fb[4];
#pragma unroll
    for (int j = 0; j < 4; ++j) {
      const __bf16* lb = &lB[buf][(wn * 64 + j * 16 + ln) * kRS];
      v8bf lo = *(const v8bf*)(lb + 8 * hf);
      v8bf hi = *(const v8bf*)(lb + 16 + 8 * hf);
      fb[j] = SHUF16(lo, hi);
    }
#pragma unroll
    for (int i = 0; i < 2; ++i)
#pragma unroll
      for (int j = 0; j < 4; ++j)
        acc[i][j] = __builtin_amdgcn_wmma_f32_16x16x32_bf16(
            false, fa[i], false, fb[j], (short)0, acc[i][j], false, false);
  }
#undef STAGE_ALL

#pragma unroll
  for (int i = 0; i < 2; ++i)
#pragma unroll
    for (int j = 0; j < 4; ++j)
      store_tile(C, Cb, N, rowBase + wm * 32 + i * 16,
                 colBase + wn * 64 + j * 16, lane, acc[i][j], bias, act);
}

// ---------------- causal depthwise conv (k=4) + SiLU (emits f32 + bf16) ----------------
__global__ __launch_bounds__(256) void dwconv_silu_kernel(
    const float* __restrict__ XZ, const float* __restrict__ cw,
    const float* __restrict__ cb, float* __restrict__ XM, __bf16* __restrict__ XMb) {
  int tid = blockIdx.x * 256 + threadIdx.x;          // BL*DI
  int d   = tid & (kDI - 1);
  int row = tid >> 10;
  int l   = row & (kL - 1);
  int base = row - l;
  float acc = cb[d];
#pragma unroll
  for (int j = 0; j < kDC; ++j) {
    int ll = l - (kDC - 1) + j;
    if (ll >= 0) acc += cw[d * kDC + j] * XZ[(size_t)(base + ll) * (2 * kDI) + d];
  }
  float v = silu_f(acc);
  XM[(size_t)row * kDI + d]  = v;
  XMb[(size_t)row * kDI + d] = f2bf(v);
}

// ---------------- selective scan, one thread per (b, d) chain ----------------
__global__ __launch_bounds__(256) void scan_kernel(
    const float* __restrict__ XM, const float* __restrict__ Delta,
    const float* __restrict__ DBC, const float* __restrict__ XZ,
    const float* __restrict__ Alog, const float* __restrict__ Dvec,
    __bf16* __restrict__ Yb) {
  int tid = blockIdx.x * 256 + threadIdx.x;   // B*DI
  int d = tid & (kDI - 1);
  int b = tid >> 10;

  float Arow[kDS];
#pragma unroll
  for (int s = 0; s < kDS; ++s) Arow[s] = -expf(Alog[d * kDS + s]);
  const float Dd = Dvec[d];

  float h[kDS];
#pragma unroll
  for (int s = 0; s < kDS; ++s) h[s] = 0.0f;

  for (int l = 0; l < kL; ++l) {
    size_t row = (size_t)b * kL + l;
    float delta = Delta[row * kDI + d];
    float u     = XM[row * kDI + d];
    float res   = XZ[row * (2 * kDI) + kDI + d];
    const float* bc = DBC + row * 96;
    float y = 0.0f;
    float du = delta * u;
#pragma unroll
    for (int s = 0; s < kDS; ++s) {
      float dA = expf(delta * Arow[s]);
      h[s] = dA * h[s] + du * bc[32 + s];
      y += h[s] * bc[64 + s];
    }
    y = (y + u * Dd) * silu_f(res);
    Yb[row * kDI + d] = f2bf(y);
  }
}

// ---------------- LayerNorm over rows of DM=512 (emits f32 + bf16) ----------------
__global__ __launch_bounds__(256) void layernorm_kernel(
    const float* __restrict__ in, const float* __restrict__ g,
    const float* __restrict__ b, float* __restrict__ out, __bf16* __restrict__ outb) {
  __shared__ float redS[8], redQ[8];
  int row = blockIdx.x;
  int t = threadIdx.x;
  const float* x = in + (size_t)row * kDM;
  float v0 = x[t], v1 = x[t + 256];
  float s = v0 + v1, q = v0 * v0 + v1 * v1;
#pragma unroll
  for (int o = 16; o > 0; o >>= 1) { s += __shfl_xor(s, o, 32); q += __shfl_xor(q, o, 32); }
  if ((t & 31) == 0) { redS[t >> 5] = s; redQ[t >> 5] = q; }
  __syncthreads();
  float ts = 0.0f, tq = 0.0f;
#pragma unroll
  for (int i = 0; i < 8; ++i) { ts += redS[i]; tq += redQ[i]; }
  float mu = ts * (1.0f / kDM);
  float var = tq * (1.0f / kDM) - mu * mu;
  float rstd = rsqrtf(var + 1e-5f);
  float o0 = (v0 - mu) * rstd * g[t]       + b[t];
  float o1 = (v1 - mu) * rstd * g[t + 256] + b[t + 256];
  size_t i0 = (size_t)row * kDM + t;
  out[i0] = o0;        out[i0 + 256] = o1;
  outb[i0] = f2bf(o0); outb[i0 + 256] = f2bf(o1);
}

// ---------------- mean pool over time ----------------
__global__ __launch_bounds__(256) void pool_kernel(
    const float* __restrict__ H, float* __restrict__ pooled) {
  int tid = blockIdx.x * 256 + threadIdx.x;   // B*DM
  int d = tid & (kDM - 1);
  int b = tid >> 9;
  float s = 0.0f;
  for (int l = 0; l < kL; ++l) s += H[((size_t)b * kL + l) * kDM + d];
  pooled[b * kDM + d] = s * (1.0f / kL);
}

// ---------------- classifier head (M=8, scalar f32) ----------------
__global__ __launch_bounds__(256) void classifier_kernel(
    const float* __restrict__ pooled, const float* __restrict__ Wout,
    const float* __restrict__ bout, float* __restrict__ out) {
  int tid = blockIdx.x * 256 + threadIdx.x;
  if (tid >= kB * kNC) return;
  int b = tid / kNC, c = tid - b * kNC;
  float s = bout[c];
  const float* p = pooled + b * kDM;
  const float* w = Wout + (size_t)c * kDM;
  for (int k = 0; k < kDM; ++k) s += p[k] * w[k];
  out[tid] = s;
}

// ---------------- launch ----------------
extern "C" void kernel_launch(void* const* d_in, const int* in_sizes, int n_in,
                              void* d_out, int out_size, void* d_ws, size_t ws_size,
                              hipStream_t stream) {
  const float* x      = (const float*)d_in[0];
  const float* Wp     = (const float*)d_in[1];
  const float* bp     = (const float*)d_in[2];
  const float* Wi     = (const float*)d_in[3];
  const float* conv_w = (const float*)d_in[4];
  const float* conv_b = (const float*)d_in[5];
  const float* Wx     = (const float*)d_in[6];
  const float* Wdt    = (const float*)d_in[7];
  const float* bdt    = (const float*)d_in[8];
  const float* A_log  = (const float*)d_in[9];
  const float* Dp     = (const float*)d_in[10];
  const float* Wo     = (const float*)d_in[11];
  const float* ln_g   = (const float*)d_in[12];
  const float* ln_b   = (const float*)d_in[13];
  const float* Wout   = (const float*)d_in[14];
  const float* bout   = (const float*)d_in[15];
  float* outp = (float*)d_out;

  size_t off = 0;
  auto carve = [&](size_t bytes) -> char* {
    char* p = (char*)d_ws + off;
    off += (bytes + 255) & ~(size_t)255;
    return p;
  };
  float*  H      = (float*) carve((size_t)kBL * kDM * 4);
  __bf16* Hb     = (__bf16*)carve((size_t)kBL * kDM * 2);
  __bf16* Xb     = (__bf16*)carve((size_t)kBL * kMels * 2);
  __bf16* Wb     = (__bf16*)carve((size_t)2 * kDI * kDM * 2);
  float*  XZ     = (float*) carve((size_t)kBL * 2 * kDI * 4);
  float*  XM     = (float*) carve((size_t)kBL * kDI * 4);
  __bf16* XMb    = (__bf16*)carve((size_t)kBL * kDI * 2);
  float*  DBC    = (float*) carve((size_t)kBL * 96 * 4);
  __bf16* DBCb   = (__bf16*)carve((size_t)kBL * 96 * 2);
  float*  Delta  = (float*) carve((size_t)kBL * kDI * 4);
  __bf16* Yb     = (__bf16*)carve((size_t)kBL * kDI * 2);
  float*  OUTf   = (float*) carve((size_t)kBL * kDM * 4);
  float*  pooled = (float*) carve((size_t)kB * kDM * 4);

  auto cvt = [&](const float* s, __bf16* d, int n) {
    f32_to_bf16_kernel<<<(n + 255) / 256, 256, 0, stream>>>(s, d, n);
  };
  auto gemmL = [&](const __bf16* A, const __bf16* W, const float* bias,
                   float* C, __bf16* Cb, int M, int N, int K, int act) {
    wmma_gemm_lds<<<dim3(M / 128, N / 128), 256, 0, stream>>>(A, W, bias, C, Cb, N, K, act);
  };
  auto gemmD = [&](const __bf16* A, int lda, const __bf16* W, const float* bias,
                   float* C, __bf16* Cb, int M, int N, int K, int act) {
    wmma_gemm_direct<<<dim3(M / 32, N / 32), 32, 0, stream>>>(A, lda, W, bias, C, Cb, N, K, act);
  };

  // ---- input projection: H(+Hb) = x @ Wp^T + bp ----
  cvt(x,  Xb, kBL * kMels);
  cvt(Wp, Wb, kDM * kMels);
  gemmL(Xb, Wb, bp, H, Hb, kBL, kDM, kMels, ACT_NONE);

  for (int i = 0; i < kNL; ++i) {
    // in_proj: XZ = H @ Wi^T   [BL, 2*DI]
    cvt(Wi + (size_t)i * 2 * kDI * kDM, Wb, 2 * kDI * kDM);
    gemmL(Hb, Wb, nullptr, XZ, nullptr, kBL, 2 * kDI, kDM, ACT_NONE);

    // causal depthwise conv + SiLU -> XM (f32) + XMb (bf16)
    dwconv_silu_kernel<<<(kBL * kDI) / 256, 256, 0, stream>>>(
        XZ, conv_w + (size_t)i * kDI * kDC, conv_b + (size_t)i * kDI, XM, XMb);

    // x_proj: DBC(+DBCb) = XM @ Wx^T   [BL, 96]
    cvt(Wx + (size_t)i * 96 * kDI, Wb, 96 * kDI);
    gemmD(XMb, kDI, Wb, nullptr, DBC, DBCb, kBL, 96, kDI, ACT_NONE);

    // dt_proj: Delta = softplus(dt @ Wdt^T + bdt)   (dt = DBCb[:, :32], lda = 96)
    cvt(Wdt + (size_t)i * kDI * kDTR, Wb, kDI * kDTR);
    gemmD(DBCb, 96, Wb, bdt + (size_t)i * kDI, Delta, nullptr, kBL, kDI, kDTR, ACT_SOFTPLUS);

    // selective scan (fuses * silu(res), writes bf16 Y)
    scan_kernel<<<(kB * kDI) / 256, 256, 0, stream>>>(
        XM, Delta, DBC, XZ, A_log + (size_t)i * kDI * kDS, Dp + (size_t)i * kDI, Yb);

    // out_proj: OUT = Y @ Wo^T  [BL, DM]
    cvt(Wo + (size_t)i * kDM * kDI, Wb, kDM * kDI);
    gemmL(Yb, Wb, nullptr, OUTf, nullptr, kBL, kDM, kDI, ACT_NONE);

    // layernorm -> H (f32) + Hb (bf16)
    layernorm_kernel<<<kBL, 256, 0, stream>>>(OUTf, ln_g + (size_t)i * kDM,
                                              ln_b + (size_t)i * kDM, H, Hb);
  }

  pool_kernel<<<(kB * kDM) / 256, 256, 0, stream>>>(H, pooled);
  classifier_kernel<<<(kB * kNC + 255) / 256, 256, 0, stream>>>(pooled, Wout, bout, outp);
}
